// GRUOdeDecay_34479997452543
// MI455X (gfx1250) — matrix operand order repaired and checked by reading, not compiled, not run on hardware
//
#include <hip/hip_runtime.h>
#include <hip/hip_bf16.h>
#include <stdint.h>

// ---------------- CDNA5 WMMA types ----------------
typedef __bf16 v16bf __attribute__((ext_vector_type(16)));
typedef float  v8f   __attribute__((ext_vector_type(8)));

#define Tt 128
#define Hh 512
#define Ee 512
#define Bb 512
#define MB 16            // batch rows per workgroup (one WMMA M tile)
#define LDH 520          // padded row stride (elements); ~2-way LDS bank sched (optimal for b128 A loads)
#define NTHREADS 512     // 16 waves (wave32)

struct alignas(16) U4 { uint32_t x, y, z, w; };

union FragB16 { uint32_t u[8]; v16bf v; };

__device__ __forceinline__ unsigned short f2bf(float f) {
    union { float f; uint32_t u; } x; x.f = f;
    uint32_t r = (x.u + 0x7FFFu + ((x.u >> 16) & 1u)) >> 16;
    return (unsigned short)r;
}

__device__ __forceinline__ float sigmoid_f(float x) { return 1.f / (1.f + expf(-x)); }
__device__ __forceinline__ float softplus_f(float x) { return (x > 15.f) ? x : log1pf(expf(x)); }

// A fragment: 16x32 bf16 tile from LDS (rows = 16 batch rows).
// ISA layout: lane(l%16)=M; lanes<16 hold K {0..7,16..23}, lanes>=16 hold K {8..15,24..31}.
__device__ __forceinline__ v16bf loadA(const unsigned short* base, int lm, int half, int k0) {
    FragB16 f;
    const U4* p0 = (const U4*)(base + lm * LDH + k0 + half * 8);
    const U4* p1 = (const U4*)(base + lm * LDH + k0 + 16 + half * 8);
    U4 a = *p0, b = *p1;
    f.u[0] = a.x; f.u[1] = a.y; f.u[2] = a.z; f.u[3] = a.w;
    f.u[4] = b.x; f.u[5] = b.y; f.u[6] = b.z; f.u[7] = b.w;
    return f.v;
}

// B fragment: 32x16 bf16 for h @ W^T: column n = W row (j0+n); lane holds
// K {k0+half*8..+7, k0+16+half*8..+7} (same K order as loadA).
__device__ __forceinline__ v16bf loadB(const unsigned short* rowbase, int half, int k0) {
    FragB16 f;
    const U4* p0 = (const U4*)(rowbase + k0 + half * 8);
    const U4* p1 = (const U4*)(rowbase + k0 + 16 + half * 8);
    U4 a = *p0, b = *p1;
    f.u[0] = a.x; f.u[1] = a.y; f.u[2] = a.z; f.u[3] = a.w;
    f.u[4] = b.x; f.u[5] = b.y; f.u[6] = b.z; f.u[7] = b.w;
    return f.v;
}

__device__ __forceinline__ v8f wmma_bf16(v16bf a, v16bf b, v8f c) {
    return __builtin_amdgcn_wmma_f32_16x16x32_bf16(false, a, false, b, (short)0, c, false, false);
}

// Software-pipelined dual-tile GEMM: acc{0,1}[16x16] += A(16x512) @ W^T over this
// wave's two column tiles. A-fragment shared by both WMMAs; next iteration's three
// fragments are loaded before the current WMMAs consume the live buffers.
__device__ __forceinline__ void gemm2_pipelined(const unsigned short* __restrict__ Sin,
                                                const unsigned short* __restrict__ pW0,
                                                const unsigned short* __restrict__ pW1,
                                                int lm, int half, v8f& acc0, v8f& acc1) {
    v16bf a  = loadA(Sin, lm, half, 0);
    v16bf b0 = loadB(pW0, half, 0);
    v16bf b1 = loadB(pW1, half, 0);
#pragma unroll 4
    for (int k0 = 0; k0 < Hh - 32; k0 += 32) {
        v16bf an  = loadA(Sin, lm, half, k0 + 32);
        v16bf b0n = loadB(pW0, half, k0 + 32);
        v16bf b1n = loadB(pW1, half, k0 + 32);
        acc0 = wmma_bf16(a, b0, acc0);
        acc1 = wmma_bf16(a, b1, acc1);
        a = an; b0 = b0n; b1 = b1n;
    }
    acc0 = wmma_bf16(a, b0, acc0);
    acc1 = wmma_bf16(a, b1, acc1);
}

// ---------------- prep kernels ----------------
__global__ void conv_bf16_kernel(const float* __restrict__ in, unsigned short* __restrict__ out, int n) {
    for (int i = blockIdx.x * blockDim.x + threadIdx.x; i < n; i += gridDim.x * blockDim.x)
        out[i] = f2bf(in[i]);
}

__global__ void tmin_kernel(const float* __restrict__ times, float* __restrict__ tmin) {
    int t = threadIdx.x;                       // 128 threads, one per time step
    float m = 3.4028235e38f;
    for (int b = 0; b < Bb; ++b) m = fminf(m, times[(size_t)b * Tt + t]);
    tmin[t] = m;
}

// ---------------- main persistent kernel ----------------
__global__ __launch_bounds__(NTHREADS, 1)
void gruode_main(const float* __restrict__ inputs, const float* __restrict__ times,
                 const float* __restrict__ bih, const float* __restrict__ bhh,
                 const float* __restrict__ b1, const float* __restrict__ b2,
                 const float* __restrict__ b3,
                 const unsigned short* __restrict__ Wihb, const unsigned short* __restrict__ Whhb,
                 const unsigned short* __restrict__ W1b, const unsigned short* __restrict__ W2b,
                 const unsigned short* __restrict__ W3b,
                 const float* __restrict__ tmin, float* __restrict__ out) {
    extern __shared__ char smem[];
    unsigned short* hs = (unsigned short*)smem;   // bf16 h / RK4 stage input
    unsigned short* xs = hs + MB * LDH;           // bf16 x tile
    unsigned short* su = xs + MB * LDH;           // MLP stage ping
    unsigned short* sv = su + MB * LDH;           // MLP stage pong
    float* hf  = (float*)(sv + MB * LDH);         // f32 h master
    float* ksm = hf + MB * LDH;                   // RK4 k-sum
    float* sdt = ksm + MB * LDH;                  // per-row dt [16]

    const int tid  = threadIdx.x;
    const int lane = tid & 31;
    const int wv   = tid >> 5;        // 0..15
    const int half = lane >> 4;
    const int lm   = lane & 15;
    const int b0   = blockIdx.x * MB;

    const int ja = (wv * 2) * 16 + lm;            // this wave's two owned columns
    const int jb = ja + 16;

    for (int i = tid; i < MB * LDH; i += NTHREADS) { hf[i] = 0.f; hs[i] = 0; }
    __syncthreads();

    // per-thread biases for the two owned columns
    const float bias_rA = bih[ja] + bhh[ja],                 bias_rB = bih[jb] + bhh[jb];
    const float bias_zA = bih[Hh + ja] + bhh[Hh + ja],       bias_zB = bih[Hh + jb] + bhh[Hh + jb];
    const float bxnA = bih[2 * Hh + ja],                     bxnB = bih[2 * Hh + jb];
    const float bhnA = bhh[2 * Hh + ja],                     bhnB = bhh[2 * Hh + jb];
    const float b1A = b1[ja], b1B = b1[jb];
    const float b2A = b2[ja], b2B = b2[jb];
    const float b3A = b3[ja], b3B = b3[jb];

    // per-thread weight row bases (constant over t)
    const unsigned short* pW1a  = W1b + (size_t)ja * Hh;
    const unsigned short* pW1bp = W1b + (size_t)jb * Hh;
    const unsigned short* pW2a  = W2b + (size_t)ja * Hh;
    const unsigned short* pW2bp = W2b + (size_t)jb * Hh;
    const unsigned short* pW3a  = W3b + (size_t)ja * Hh;
    const unsigned short* pW3bp = W3b + (size_t)jb * Hh;

    for (int t = 0; t < Tt; ++t) {
        // ---- stage x tile (f32 -> bf16), per-row dt ----
        for (int i = tid; i < MB * Ee; i += NTHREADS) {
            int r = i >> 9, c = i & (Ee - 1);
            xs[r * LDH + c] = f2bf(inputs[(size_t)(b0 + r) * Tt * Ee + (size_t)t * Ee + c]);
        }
        if (tid < MB) sdt[tid] = (times[(size_t)(b0 + tid) * Tt + t] - tmin[t]) * 0.25f;
        __syncthreads();

        // ---- GRU: 6 fused gate GEMM fragments per output tile (spill-free form) ----
        float hn[2][8];
#pragma unroll
        for (int tile = 0; tile < 2; ++tile) {
            int j = (tile == 0) ? ja : jb;
            const unsigned short* pWr = Wihb + (size_t)j * Ee;
            const unsigned short* pWz = Wihb + (size_t)(Hh + j) * Ee;
            const unsigned short* pWn = Wihb + (size_t)(2 * Hh + j) * Ee;
            const unsigned short* pUr = Whhb + (size_t)j * Hh;
            const unsigned short* pUz = Whhb + (size_t)(Hh + j) * Hh;
            const unsigned short* pUn = Whhb + (size_t)(2 * Hh + j) * Hh;
            v8f arx = {}, azx = {}, anx = {}, arh = {}, azh = {}, anh = {};
            for (int k0 = 0; k0 < Ee; k0 += 32) {
                v16bf ax = loadA(xs, lm, half, k0);
                v16bf ah = loadA(hs, lm, half, k0);
                arx = wmma_bf16(ax, loadB(pWr, half, k0), arx);
                azx = wmma_bf16(ax, loadB(pWz, half, k0), azx);
                anx = wmma_bf16(ax, loadB(pWn, half, k0), anx);
                arh = wmma_bf16(ah, loadB(pUr, half, k0), arh);
                azh = wmma_bf16(ah, loadB(pUz, half, k0), azh);
                anh = wmma_bf16(ah, loadB(pUn, half, k0), anh);
            }
            float brv = (tile == 0) ? bias_rA : bias_rB;
            float bzv = (tile == 0) ? bias_zA : bias_zB;
            float bxn = (tile == 0) ? bxnA : bxnB;
            float bhn = (tile == 0) ? bhnA : bhnB;
#pragma unroll
            for (int i = 0; i < 8; ++i) {
                int M = i + 8 * half;   // C layout: VGPR i -> M = i + 8*(lane>=16)
                float r = sigmoid_f(arx[i] + arh[i] + brv);
                float z = sigmoid_f(azx[i] + azh[i] + bzv);
                float n = tanhf(anx[i] + bxn + r * (anh[i] + bhn));
                float ho = hf[M * LDH + j];
                hn[tile][i] = (1.f - z) * n + z * ho;
            }
        }
        __syncthreads();   // all waves done reading old hs before rewrite
#pragma unroll
        for (int tile = 0; tile < 2; ++tile) {
            int j = (tile == 0) ? ja : jb;
#pragma unroll
            for (int i = 0; i < 8; ++i) {
                int M = i + 8 * half;
                float v = hn[tile][i];
                hf[M * LDH + j] = v;
                hs[M * LDH + j] = f2bf(v);
                out[(size_t)(b0 + M) * Tt * Hh + (size_t)t * Hh + j] = v;  // out_t = hn
            }
        }
        __syncthreads();

        // ---- RK4: 4 iterations x (k1..k4); MLP = 3 pipelined dual-tile GEMMs ----
        for (int it = 0; it < 4; ++it) {
#pragma unroll
            for (int st = 0; st < 4; ++st) {
                // layer 1: su = softplus(hs @ W1^T + b1)
                {
                    v8f a0 = {}, a1 = {};
                    gemm2_pipelined(hs, pW1a, pW1bp, lm, half, a0, a1);
#pragma unroll
                    for (int i = 0; i < 8; ++i) {
                        int M = i + 8 * half;
                        su[M * LDH + ja] = f2bf(softplus_f(a0[i] + b1A));
                        su[M * LDH + jb] = f2bf(softplus_f(a1[i] + b1B));
                    }
                }
                __syncthreads();
                // layer 2: sv = softplus(su @ W2^T + b2)
                {
                    v8f a0 = {}, a1 = {};
                    gemm2_pipelined(su, pW2a, pW2bp, lm, half, a0, a1);
#pragma unroll
                    for (int i = 0; i < 8; ++i) {
                        int M = i + 8 * half;
                        sv[M * LDH + ja] = f2bf(softplus_f(a0[i] + b2A));
                        sv[M * LDH + jb] = f2bf(softplus_f(a1[i] + b2B));
                    }
                }
                __syncthreads();
                // layer 3: k = sv @ W3^T + b3; fold into ksum + next stage input
                {
                    v8f a0 = {}, a1 = {};
                    gemm2_pipelined(sv, pW3a, pW3bp, lm, half, a0, a1);
#pragma unroll
                    for (int tile = 0; tile < 2; ++tile) {
                        int j = (tile == 0) ? ja : jb;
                        float b3v = (tile == 0) ? b3A : b3B;
#pragma unroll
                        for (int i = 0; i < 8; ++i) {
                            int M = i + 8 * half;
                            float kk = ((tile == 0) ? a0[i] : a1[i]) + b3v;
                            float dt = sdt[M];
                            float h0 = hf[M * LDH + j];
                            if (st == 0) {
                                ksm[M * LDH + j] = kk;
                                hs[M * LDH + j] = f2bf(h0 + 0.5f * dt * kk);
                            } else if (st == 1) {
                                ksm[M * LDH + j] += 2.f * kk;
                                hs[M * LDH + j] = f2bf(h0 + 0.5f * dt * kk);
                            } else if (st == 2) {
                                ksm[M * LDH + j] += 2.f * kk;
                                hs[M * LDH + j] = f2bf(h0 + dt * kk);
                            } else {
                                float hnew = h0 + (dt * (1.f / 6.f)) * (ksm[M * LDH + j] + kk);
                                hf[M * LDH + j] = hnew;
                                hs[M * LDH + j] = f2bf(hnew);
                            }
                        }
                    }
                }
                __syncthreads();
            }
        }
    }
}

// ---------------- launch ----------------
extern "C" void kernel_launch(void* const* d_in, const int* in_sizes, int n_in,
                              void* d_out, int out_size, void* d_ws, size_t ws_size,
                              hipStream_t stream) {
    const float* inputs = (const float*)d_in[0];
    const float* times  = (const float*)d_in[1];
    const float* Wih    = (const float*)d_in[2];
    const float* Whh    = (const float*)d_in[3];
    const float* bih    = (const float*)d_in[4];
    const float* bhh    = (const float*)d_in[5];
    const float* W1     = (const float*)d_in[6];
    const float* b1     = (const float*)d_in[7];
    const float* W2     = (const float*)d_in[8];
    const float* b2     = (const float*)d_in[9];
    const float* W3     = (const float*)d_in[10];
    const float* b3     = (const float*)d_in[11];
    float* out = (float*)d_out;

    // workspace: bf16 weights (L2-resident, ~4.7 MB) + tmin[T]
    const size_t nWih = (size_t)3 * Hh * Ee;
    const size_t nWhh = (size_t)3 * Hh * Hh;
    const size_t nW   = (size_t)Hh * Hh;
    unsigned short* Wihb = (unsigned short*)d_ws;
    unsigned short* Whhb = Wihb + nWih;
    unsigned short* W1b  = Whhb + nWhh;
    unsigned short* W2b  = W1b + nW;
    unsigned short* W3b  = W2b + nW;
    float* tminp = (float*)(W3b + nW);

    conv_bf16_kernel<<<512, 256, 0, stream>>>(Wih, Wihb, (int)nWih);
    conv_bf16_kernel<<<512, 256, 0, stream>>>(Whh, Whhb, (int)nWhh);
    conv_bf16_kernel<<<256, 256, 0, stream>>>(W1, W1b, (int)nW);
    conv_bf16_kernel<<<256, 256, 0, stream>>>(W2, W2b, (int)nW);
    conv_bf16_kernel<<<256, 256, 0, stream>>>(W3, W3b, (int)nW);
    tmin_kernel<<<1, Tt, 0, stream>>>(times, tminp);

    const int SMEM = 4 * MB * LDH * 2   // 4 bf16 buffers
                   + 2 * MB * LDH * 4   // hf + ksum (f32)
                   + 16 * 4;            // sdt
    (void)hipFuncSetAttribute((const void*)gruode_main,
                              hipFuncAttributeMaxDynamicSharedMemorySize, SMEM);
    gruode_main<<<Bb / MB, NTHREADS, SMEM, stream>>>(
        inputs, times, bih, bhh, b1, b2, b3,
        Wihb, Whhb, W1b, W2b, W3b, tminp, out);
}
